// CausalSelfAttention_50697793962699
// MI455X (gfx1250) — matrix operand order repaired and checked by reading, not compiled, not run on hardware
//
#include <hip/hip_runtime.h>
#include <hip/hip_bf16.h>

// ---------------- problem constants ----------------
#define NE   768           // n_embd
#define HN   12            // heads
#define HD   64            // head dim
#define LQ   2048          // sequence length
#define BB   2             // batch
#define BL   (BB * LQ)     // 4096 tokens
#define QKV3 (3 * NE)      // 2304

// ---------------- WMMA types ----------------
typedef __attribute__((ext_vector_type(16))) __bf16 bf16x16;
typedef __attribute__((ext_vector_type(8)))  float  floatx8;

// A fragment (16x32 bf16, MxK, wave32): lane<16 -> row M=lane, K elements
// {0..7,16..23}+kbase ; lane>=16 -> row M=lane-16, those +8. Two 16B chunks.
__device__ __forceinline__ bf16x16 load_a_frag(const __bf16* rowbase, int kbase, int lane) {
    const int ko = kbase + ((lane < 16) ? 0 : 8);
    union { bf16x16 v; uint4 q[2]; } f;
    f.q[0] = *(const uint4*)(rowbase + ko);
    f.q[1] = *(const uint4*)(rowbase + ko + 16);
    return f.v;
}

// B fragment (32x16 bf16, KxN): lane<16 -> col lane, K elements
// {0..7,16..23}+kbase ; lane>=16 -> col lane-16, those +8. Two 16B chunks.
__device__ __forceinline__ bf16x16 load_b_frag(const __bf16* colbase, int kbase, int lane) {
    const int ko = kbase + ((lane < 16) ? 0 : 8);
    union { bf16x16 v; uint4 q[2]; } f;
    f.q[0] = *(const uint4*)(colbase + ko);
    f.q[1] = *(const uint4*)(colbase + ko + 16);
    return f.v;
}

__device__ __forceinline__ floatx8 wmma_bf16(bf16x16 a, bf16x16 b, floatx8 c) {
    return __builtin_amdgcn_wmma_f32_16x16x32_bf16(false, a, false, b, (short)0, c,
                                                   false, false);
}

__device__ __forceinline__ unsigned int pack_bf16(float lo, float hi) {
    unsigned short l = __builtin_bit_cast(unsigned short, (__bf16)lo);
    unsigned short h = __builtin_bit_cast(unsigned short, (__bf16)hi);
    return (unsigned int)l | ((unsigned int)h << 16);
}

// ---- CDNA5 async copy: 16B global -> LDS, tracked by ASYNCcnt ----
// Generic LDS pointer: low 32 bits are the wave-relative LDS byte address.
__device__ __forceinline__ void async_copy16(const void* g, const void* l) {
    unsigned lo = (unsigned)(uintptr_t)l;
    asm volatile("global_load_async_to_lds_b128 %0, %1, off"
                 :: "v"(lo), "v"(g) : "memory");
}
__device__ __forceinline__ void async_wait0() {
    asm volatile("s_wait_asynccnt 0x0" ::: "memory");
}

// =====================================================================
// bf16 GEMM: C[M,N] = A[M,K] * BT[N,K]^T + bias.  A,BT bf16; bias f32.
// store_bf: 1 -> bf16 output, 0 -> f32 output (+tanh if act).
// qscale: scale columns with (n%192)<64 by 0.125 (folds softmax scale into Q).
// Workgroup tile 128x64 (4 waves, each 32 rows x 64 cols). K-step 32,
// double-buffered LDS filled with async b128 copies.
// =====================================================================
__global__ __launch_bounds__(128) void gemm_bf16_kernel(
    const __bf16* __restrict__ A, const __bf16* __restrict__ BT,
    const float* __restrict__ bias, void* __restrict__ Cout,
    int M, int N, int K, int act, int store_bf, int qscale)
{
    __shared__ __bf16 As[2][128][40];   // [m][k], 80B row stride (16B aligned)
    __shared__ __bf16 Bs[2][64][40];    // [n][k]

    const int tid  = threadIdx.x;
    const int lane = tid & 31;
    const int wave = tid >> 5;
    const int col  = lane & 15;
    const int n0   = blockIdx.x * 64;
    const int m0   = blockIdx.y * 128;

    auto issue = [&](int buf, int k0) {
#pragma unroll
        for (int t = 0; t < 4; ++t) {           // A tile: 128 rows x 4 chunks
            int c_ = tid + t * 128;
            int row = c_ >> 2, ch = (c_ & 3) * 8;
            async_copy16(A + (size_t)(m0 + row) * K + k0 + ch, &As[buf][row][ch]);
        }
#pragma unroll
        for (int t = 0; t < 2; ++t) {           // B tile: 64 rows x 4 chunks
            int c_ = tid + t * 128;
            int n = c_ >> 2, ch = (c_ & 3) * 8;
            async_copy16(BT + (size_t)(n0 + n) * K + k0 + ch, &Bs[buf][n][ch]);
        }
    };

    floatx8 acc[2][4] = {{{0,0,0,0,0,0,0,0},{0,0,0,0,0,0,0,0},
                          {0,0,0,0,0,0,0,0},{0,0,0,0,0,0,0,0}},
                         {{0,0,0,0,0,0,0,0},{0,0,0,0,0,0,0,0},
                          {0,0,0,0,0,0,0,0},{0,0,0,0,0,0,0,0}}};
    const int T = K / 32;
    issue(0, 0);
    for (int t = 0; t < T; ++t) {
        async_wait0();
        __syncthreads();                        // tile t visible to all waves
        if (t + 1 < T) issue((t + 1) & 1, (t + 1) * 32);
        const int buf = t & 1;
        bf16x16 a0 = load_a_frag(&As[buf][wave * 32 + col][0], 0, lane);
        bf16x16 a1 = load_a_frag(&As[buf][wave * 32 + 16 + col][0], 0, lane);
#pragma unroll
        for (int c = 0; c < 4; ++c) {
            bf16x16 b = load_b_frag(&Bs[buf][c * 16 + col][0], 0, lane);
            acc[0][c] = wmma_bf16(a0, b, acc[0][c]);
            acc[1][c] = wmma_bf16(a1, b, acc[1][c]);
        }
    }

    // Epilogue: acc element i -> row (i | i+8), col = lane&15
    const int rb = (lane < 16) ? 0 : 8;
    float* Cf = (float*)Cout;
    __bf16* Cb = (__bf16*)Cout;
#pragma unroll
    for (int mi = 0; mi < 2; ++mi)
#pragma unroll
        for (int c = 0; c < 4; ++c) {
            int n = n0 + c * 16 + col;
            float bv = bias[n];
            float scl = (qscale && ((n % 192) < 64)) ? 0.125f : 1.0f;
#pragma unroll
            for (int i = 0; i < 8; ++i) {
                float v = (acc[mi][c][i] + bv) * scl;
                if (act) v = tanhf(v);
                size_t idx = (size_t)(m0 + wave * 32 + mi * 16 + rb + i) * N + n;
                if (store_bf) Cb[idx] = (__bf16)v; else Cf[idx] = v;
            }
        }
}

// =====================================================================
// Flash attention (causal), bf16 inputs. qkv[token][2304] with head h at
// cols [h*192 .. +64)=Q(prescaled by 0.125), +64..128=K, +128..192=V.
// vT[bh][d][token] is the pre-transposed V. Output y bf16 [token][768].
// Grid (L/64, B*H), 4 waves; wave w owns query rows w*16..+15.
// K/V tiles (64 keys) double-buffered via async b128 copies.
// =====================================================================
__global__ __launch_bounds__(128) void attn_kernel(
    const __bf16* __restrict__ qkv, const __bf16* __restrict__ vT,
    __bf16* __restrict__ y)
{
    __shared__ __bf16 Ks[2][64][72];     // [key][d]  (B frag for S=Q*K^T)
    __shared__ __bf16 Vs[2][64][72];     // [d][key]  (B frag for O=P*V)
    __shared__ float  Ss[4][16][68];     // per-wave scores f32
    __shared__ __bf16 Ps[4][16][72];     // per-wave probs bf16 (A frag)

    const int tid  = threadIdx.x;
    const int lane = tid & 31;
    const int wave = tid >> 5;
    const int col  = lane & 15;
    const int bh   = blockIdx.y;
    const int b    = bh / HN;
    const int h    = bh % HN;
    const int q0   = blockIdx.x * 64;
    const size_t tokbase = (size_t)b * LQ;
    const int qoff = h * 3 * HD;

    // Q fragments (rows = wave's 16 queries), direct b128 global loads
    const int qrow = wave * 16 + col;
    const __bf16* qp = qkv + (tokbase + q0 + qrow) * (size_t)QKV3 + qoff;
    bf16x16 qf[2];
#pragma unroll
    for (int s = 0; s < 2; ++s) {
        const int base = s * 32 + ((lane < 16) ? 0 : 8);
        union { bf16x16 v; uint4 q[2]; } f;
        f.q[0] = *(const uint4*)(qp + base);
        f.q[1] = *(const uint4*)(qp + base + 16);
        qf[s] = f.v;
    }

    auto issue = [&](int buf, int key0) {
#pragma unroll
        for (int t = 0; t < 4; ++t) {           // K tile: 64 keys x 8 chunks
            int c_ = tid + t * 128;
            int r = c_ >> 3, ch = (c_ & 7) * 8;
            async_copy16(qkv + (tokbase + key0 + r) * (size_t)QKV3 + qoff + HD + ch,
                         &Ks[buf][r][ch]);
        }
        const __bf16* vbase = vT + (size_t)bh * HD * LQ;
#pragma unroll
        for (int t = 0; t < 4; ++t) {           // V^T tile: 64 d x 8 chunks
            int c_ = tid + t * 128;
            int d = c_ >> 3, ch = (c_ & 7) * 8;
            async_copy16(vbase + (size_t)d * LQ + key0 + ch, &Vs[buf][d][ch]);
        }
    };

    float m = -__builtin_inff();
    float l = 0.0f;
    floatx8 o[4] = {{0,0,0,0,0,0,0,0},{0,0,0,0,0,0,0,0},
                    {0,0,0,0,0,0,0,0},{0,0,0,0,0,0,0,0}};
    const int ntiles = q0 / 64 + 1;
    const int rb = (lane < 16) ? 0 : 8;
    const int qwmax = q0 + wave * 16 + 15;

    issue(0, 0);
    for (int kt = 0; kt < ntiles; ++kt) {
        const int key0 = kt * 64;
        async_wait0();
        __syncthreads();                         // tile kt ready for all waves
        if (kt + 1 < ntiles) issue((kt + 1) & 1, (kt + 1) * 64);
        const int buf = kt & 1;

        if (key0 <= qwmax) {                     // wave-uniform causal skip
            // ---- S = Q * K^T  (8 wmma)
            floatx8 s[4] = {{0,0,0,0,0,0,0,0},{0,0,0,0,0,0,0,0},
                            {0,0,0,0,0,0,0,0},{0,0,0,0,0,0,0,0}};
#pragma unroll
            for (int st = 0; st < 2; ++st)
#pragma unroll
                for (int c = 0; c < 4; ++c) {
                    bf16x16 bk = load_b_frag(&Ks[buf][c * 16 + col][0], st * 32, lane);
                    s[c] = wmma_bf16(qf[st], bk, s[c]);
                }
            // spill scores to LDS for row-wise softmax
#pragma unroll
            for (int c = 0; c < 4; ++c)
#pragma unroll
                for (int i = 0; i < 8; ++i)
                    Ss[wave][rb + i][c * 16 + col] = s[c][i];
            asm volatile("s_wait_dscnt 0x0" ::: "memory");

            // ---- online softmax (lane r<16 owns row r); Q already prescaled
            float mtile = -3.0e38f;
            if (lane < 16) {
                const int qi = q0 + wave * 16 + lane;
                float* srow = &Ss[wave][lane][0];
                for (int c = 0; c < 64; c += 4) {
                    float4 sv = *(float4*)(srow + c);
                    sv.x = (key0 + c     <= qi) ? sv.x : -3.0e38f;
                    sv.y = (key0 + c + 1 <= qi) ? sv.y : -3.0e38f;
                    sv.z = (key0 + c + 2 <= qi) ? sv.z : -3.0e38f;
                    sv.w = (key0 + c + 3 <= qi) ? sv.w : -3.0e38f;
                    *(float4*)(srow + c) = sv;
                    mtile = fmaxf(fmaxf(mtile, fmaxf(sv.x, sv.y)), fmaxf(sv.z, sv.w));
                }
            }
            float mnew  = fmaxf(m, mtile);
            float alpha = __expf(m - mnew);
            float psum  = 0.0f;
            if (lane < 16) {
                float* srow = &Ss[wave][lane][0];
                __bf16* prow = &Ps[wave][lane][0];
                for (int c = 0; c < 64; c += 4) {
                    float4 sv = *(float4*)(srow + c);
                    float p0 = __expf(sv.x - mnew), p1 = __expf(sv.y - mnew);
                    float p2 = __expf(sv.z - mnew), p3 = __expf(sv.w - mnew);
                    uint2 pk; pk.x = pack_bf16(p0, p1); pk.y = pack_bf16(p2, p3);
                    *(uint2*)(prow + c) = pk;
                    psum += (p0 + p1) + (p2 + p3);
                }
            }
            l = l * alpha + psum;
            m = mnew;

            // ---- rescale O by per-row alpha (broadcast from lane==row)
#pragma unroll
            for (int i = 0; i < 8; ++i) {
                float ai = __shfl(alpha, rb + i, 32);
#pragma unroll
                for (int dc = 0; dc < 4; ++dc) o[dc][i] *= ai;
            }

            // ---- O += P * V  (8 wmma)
            asm volatile("s_wait_dscnt 0x0" ::: "memory");
#pragma unroll
            for (int kc = 0; kc < 2; ++kc) {
                bf16x16 pa = load_a_frag(&Ps[wave][col][0], kc * 32, lane);
#pragma unroll
                for (int dc = 0; dc < 4; ++dc) {
                    bf16x16 bv = load_b_frag(&Vs[buf][dc * 16 + col][0], kc * 32, lane);
                    o[dc] = wmma_bf16(pa, bv, o[dc]);
                }
            }
        }
    }

    // ---- finalize: divide by l (per-row broadcast), store bf16 y
    float linv = (lane < 16 && l > 0.0f) ? 1.0f / l : 0.0f;
#pragma unroll
    for (int i = 0; i < 8; ++i) {
        float li = __shfl(linv, rb + i, 32);
        const size_t tok = tokbase + q0 + wave * 16 + rb + i;
#pragma unroll
        for (int dc = 0; dc < 4; ++dc)
            y[tok * NE + h * HD + dc * 16 + col] = (__bf16)(o[dc][i] * li);
    }
}

// =====================================================================
// prep kernels (one-time, bandwidth-trivial on 23.3 TB/s HBM)
// =====================================================================
__global__ void cvt_bf16_kernel(const float* __restrict__ in,
                                __bf16* __restrict__ out, int n) {
    int i = blockIdx.x * 256 + threadIdx.x;
    if (i < n) out[i] = (__bf16)in[i];
}

// out[n][k] = in[k][n]  (weights transposed so GEMM B tiles are raw copies)
__global__ void transpose_cvt_kernel(const float* __restrict__ in,
                                     __bf16* __restrict__ out, int K, int N) {
    int i = blockIdx.x * 256 + threadIdx.x;
    if (i < K * N) {
        int n = i / K, k = i % K;
        out[i] = (__bf16)in[(size_t)k * N + n];
    }
}

// vT[bh][d][tok] = qkv[b*LQ+tok][h*192+128+d]
__global__ void build_vT_kernel(const __bf16* __restrict__ qkv,
                                __bf16* __restrict__ vT) {
    int i = blockIdx.x * 256 + threadIdx.x;     // over BB*HN*HD*LQ
    int tok = i & (LQ - 1);
    int d   = (i >> 11) & (HD - 1);
    int bh  = i >> 17;
    int b = bh / HN, h = bh % HN;
    vT[i] = qkv[((size_t)b * LQ + tok) * QKV3 + h * 3 * HD + 2 * HD + d];
}

// =====================================================================
// launch
// =====================================================================
extern "C" void kernel_launch(void* const* d_in, const int* in_sizes, int n_in,
                              void* d_out, int out_size, void* d_ws, size_t ws_size,
                              hipStream_t stream) {
    const float* x    = (const float*)d_in[0];
    const float* Wqkv = (const float*)d_in[1];
    const float* bqkv = (const float*)d_in[2];
    const float* Wo   = (const float*)d_in[3];
    const float* bo   = (const float*)d_in[4];

    __bf16* x_bf   = (__bf16*)d_ws;                       // [BL][NE]
    __bf16* WqkvT  = x_bf   + (size_t)BL * NE;            // [QKV3][NE]
    __bf16* WoT    = WqkvT  + (size_t)QKV3 * NE;          // [NE][NE]
    __bf16* qkv_bf = WoT    + (size_t)NE * NE;            // [BL][QKV3]
    __bf16* vT     = qkv_bf + (size_t)BL * QKV3;          // [BB*HN][HD][LQ]
    __bf16* y_bf   = vT     + (size_t)BB * HN * HD * LQ;  // [BL][NE]

    dim3 blk(128), cblk(256);

    // prep: bf16 copies / transposes
    cvt_bf16_kernel<<<(BL * NE + 255) / 256, cblk, 0, stream>>>(x, x_bf, BL * NE);
    transpose_cvt_kernel<<<(NE * QKV3 + 255) / 256, cblk, 0, stream>>>(Wqkv, WqkvT, NE, QKV3);
    transpose_cvt_kernel<<<(NE * NE + 255) / 256, cblk, 0, stream>>>(Wo, WoT, NE, NE);

    // 1) qkv = x @ Wqkv + bqkv  (Q columns prescaled by 1/sqrt(D)), bf16 out
    gemm_bf16_kernel<<<dim3(QKV3 / 64, BL / 128), blk, 0, stream>>>(
        x_bf, WqkvT, bqkv, qkv_bf, BL, QKV3, NE, /*act*/0, /*store_bf*/1, /*qscale*/1);

    // 1b) pre-transpose V per head for async-copyable PV tiles
    build_vT_kernel<<<(BB * HN * HD * LQ + 255) / 256, cblk, 0, stream>>>(qkv_bf, vT);

    // 2) flash attention -> y (bf16)
    attn_kernel<<<dim3(LQ / 64, BB * HN), blk, 0, stream>>>(qkv_bf, vT, y_bf);

    // 3) out = tanh(y @ Wo + bo), f32 out
    gemm_bf16_kernel<<<dim3(NE / 64, BL / 128), blk, 0, stream>>>(
        y_bf, WoT, bo, d_out, BL, NE, NE, /*act*/1, /*store_bf*/0, /*qscale*/0);
}
